// MultiHeadAttention_42923903156436
// MI455X (gfx1250) — matrix operand order repaired and checked by reading, not compiled, not run on hardware
//
#include <hip/hip_runtime.h>

// ---------- types ----------
typedef __attribute__((ext_vector_type(16))) __bf16 v16bf;
typedef __attribute__((ext_vector_type(8)))  __bf16 v8bf;
typedef __attribute__((ext_vector_type(8)))  float  v8f;
typedef __attribute__((ext_vector_type(4)))  unsigned u32x4;
typedef __attribute__((ext_vector_type(8)))  int      i32x8;
typedef __attribute__((ext_vector_type(4)))  int      i32x4;

#define D_MODEL 1024
#define NHEAD   16
#define HDIM    64
#define BB      2
#define LL      2048
#define NTOK    (BB*LL)          // 4096 rows

static __device__ __forceinline__ __bf16 f2bf(float f) {
    union { float f; unsigned u; } v; v.f = f;
    unsigned r = v.u + 0x7fffu + ((v.u >> 16) & 1u);   // RNE
    unsigned short h = (unsigned short)(r >> 16);
    __bf16 b; __builtin_memcpy(&b, &h, 2); return b;
}

// ---------- TDM 2D tile load: global bf16 tensor -> LDS (packed row-major) ----------
// tensor_d0: row length (elems), tensor_d1: #rows, tile_d0 x tile_d1 tile,
// stride0: elems between consecutive rows.
static __device__ __forceinline__ void tdm_load_2d_bf16(
    unsigned lds_addr, const __bf16* gptr,
    unsigned tensor_d0, unsigned tensor_d1,
    unsigned tile_d0, unsigned tile_d1, unsigned stride0)
{
    unsigned long long ga = (unsigned long long)(uintptr_t)gptr;
    u32x4 g0;
    g0[0] = 1u;                                            // count=1 (valid), flags 0
    g0[1] = lds_addr;                                      // LDS byte address
    g0[2] = (unsigned)ga;                                  // global_addr[31:0]
    g0[3] = (unsigned)((ga >> 32) & 0x01FFFFFFu) | (2u << 30); // addr[56:32], type=2
    i32x8 g1;
    g1[0] = (int)(1u << 16);                               // data_size=1 -> 2 bytes
    g1[1] = (int)((tensor_d0 & 0xffffu) << 16);            // tensor_dim0[15:0] @ bit48
    g1[2] = (int)((tensor_d0 >> 16) | ((tensor_d1 & 0xffffu) << 16));
    g1[3] = (int)((tensor_d1 >> 16) | (tile_d0 << 16));    // tile_dim0 @ bit112
    g1[4] = (int)(tile_d1 & 0xffffu);                      // tile_dim1 @ bit128
    g1[5] = (int)stride0;                                  // tensor_dim0_stride[31:0]
    g1[6] = 0;
    g1[7] = 0;
    i32x4 z4 = {0, 0, 0, 0};
#if defined(__clang_major__) && (__clang_major__ >= 23)
    i32x8 z8 = {0, 0, 0, 0, 0, 0, 0, 0};
    __builtin_amdgcn_tensor_load_to_lds(g0, g1, z4, z4, z8, 0);
#else
    __builtin_amdgcn_tensor_load_to_lds(g0, g1, z4, z4, 0);
#endif
}

// ---------- cast fp32 -> bf16 ----------
__global__ void cast_f32_bf16(const float* __restrict__ in, __bf16* __restrict__ out, int n) {
    int i = blockIdx.x * blockDim.x + threadIdx.x;
    if (i < n) out[i] = f2bf(in[i]);
}

// ---------- NT GEMM: out[n,m] = sum_k A[n,k]*W[m,k] + bias[m] ----------
// A: [4096,1024] bf16 row-major, W: [1024,1024] bf16 row-major.
// W tiles staged through TDM into LDS, double buffered.
// mode 0: fp32 out[n*1024+m]
// mode 1: bf16 split-heads [B,H,L,Dh]
// mode 2: bf16 split-heads transposed [B,H,Dh,L]
__global__ __launch_bounds__(128)
void gemm_nt_bf16(const __bf16* __restrict__ A, const __bf16* __restrict__ W,
                  const float* __restrict__ bias, void* __restrict__ out, int mode) {
    __shared__ __align__(128) __bf16 wst[2][128 * 32];   // [colrel][k] per buffer

    const int lane = threadIdx.x & 31;
    const int wave = threadIdx.x >> 5;           // 4 waves
    const int l16  = lane & 15;
    const int lh   = lane >> 4;                  // 0/1 half
    const int wn   = wave & 1;
    const int row0 = blockIdx.y * 128 + (wave >> 1) * 64;
    const int colblk = blockIdx.x * 128;
    const int col0 = colblk + wn * 64;

    const unsigned wlds[2] = { (unsigned)(uintptr_t)&wst[0][0],
                               (unsigned)(uintptr_t)&wst[1][0] };

    // prologue: stage W tile for k0=0
    if (threadIdx.x < 32)
        tdm_load_2d_bf16(wlds[0], W + (size_t)colblk * D_MODEL, 32, 128, 32, 128, D_MODEL);

    v8f acc[4][4] = {};

    const int nIter = D_MODEL / 32;              // 32
    for (int it = 0; it < nIter; ++it) {
        const int k0 = it * 32;
        if (threadIdx.x < 32) {
            if (it + 1 < nIter) {
                tdm_load_2d_bf16(wlds[(it + 1) & 1],
                                 W + (size_t)colblk * D_MODEL + (k0 + 32),
                                 32, 128, 32, 128, D_MODEL);
                __builtin_amdgcn_s_wait_tensorcnt(1);
            } else {
                __builtin_amdgcn_s_wait_tensorcnt(0);
            }
        }
        __syncthreads();

        const __bf16* wbuf = &wst[it & 1][0];
        v16bf af[4], bf[4];
        const __bf16* pa0 = A + (size_t)(row0 + l16) * D_MODEL + k0 + lh * 16;
        __builtin_prefetch(pa0 + 128, 0, 0);
#pragma unroll
        for (int mi = 0; mi < 4; mi++) {
            const __bf16* pa = pa0 + (size_t)(mi * 16) * D_MODEL;
            *(v8bf*)&af[mi]       = *(const v8bf*)pa;        // k {0..7}(+16 hi)
            *((v8bf*)&af[mi] + 1) = *(const v8bf*)(pa + 8);  // k {8..15}(+16 hi)
        }
#pragma unroll
        for (int ni = 0; ni < 4; ni++) {
            const __bf16* pb = wbuf + (size_t)(wn * 64 + ni * 16 + l16) * 32 + lh * 16;
            bf[ni] = *(const v16bf*)pb;                      // k {0..15}/{16..31}
        }
#pragma unroll
        for (int mi = 0; mi < 4; mi++)
#pragma unroll
            for (int ni = 0; ni < 4; ni++)
                acc[mi][ni] = __builtin_amdgcn_wmma_f32_16x16x32_bf16(
                    false, af[mi], false, bf[ni], (short)0, acc[mi][ni], false, false);
        __syncthreads();
    }

#pragma unroll
    for (int mi = 0; mi < 4; mi++)
#pragma unroll
        for (int ni = 0; ni < 4; ni++) {
            const int coln = col0 + ni * 16 + l16;
            const float bv = bias[coln];
#pragma unroll
            for (int r = 0; r < 8; r++) {
                const int rown = row0 + mi * 16 + r + 8 * lh;
                const float v = acc[mi][ni][r] + bv;
                if (mode == 0) {
                    ((float*)out)[(size_t)rown * D_MODEL + coln] = v;
                } else {
                    const int b = rown >> 11, l = rown & (LL - 1);
                    const int h = coln >> 6,  d = coln & (HDIM - 1);
                    __bf16 bvv = f2bf(v);
                    if (mode == 1)
                        ((__bf16*)out)[(((size_t)b * NHEAD + h) * LL + l) * HDIM + d] = bvv;
                    else
                        ((__bf16*)out)[(((size_t)b * NHEAD + h) * HDIM + d) * LL + l] = bvv;
                }
            }
        }
}

// ---------- fused attention (flash-style, one wave per 16 query rows) ----------
// All 4 waves of a block share (b,h): K/V tiles staged once per block via TDM.
// Qh,Kh: [B,H,L,64] bf16; Vt: [B,H,64,L] bf16; ctx: [B*L, 1024] bf16 (merged heads)
__global__ __launch_bounds__(128)
void attn_kernel(const __bf16* __restrict__ Qh, const __bf16* __restrict__ Kh,
                 const __bf16* __restrict__ Vt, __bf16* __restrict__ ctx) {
    __shared__ __align__(128) __bf16 kst[2][32 * 64];  // [key][d]
    __shared__ __align__(128) __bf16 vst[2][64 * 32];  // [d][key]
    __shared__ __align__(32)  __bf16 pbuf[4][16 * 32];

    const int lane = threadIdx.x & 31;
    const int wave = threadIdx.x >> 5;
    const int l16  = lane & 15;
    const int lh   = lane >> 4;

    const int job  = blockIdx.x * 4 + wave;      // 0..4095
    const int qblk = job & 127;                  // 128 q-blocks per (b,h)
    const int bh   = job >> 7;                   // 0..31 (same for all 4 waves)
    const int q0   = qblk * 16;

    const __bf16* Qp = Qh + (size_t)bh * LL * HDIM;
    const __bf16* Kp = Kh + (size_t)bh * LL * HDIM;
    const __bf16* Vp = Vt + (size_t)bh * HDIM * LL;

    const unsigned klds[2] = { (unsigned)(uintptr_t)&kst[0][0],
                               (unsigned)(uintptr_t)&kst[1][0] };
    const unsigned vlds[2] = { (unsigned)(uintptr_t)&vst[0][0],
                               (unsigned)(uintptr_t)&vst[1][0] };

    // prologue: stage K/V tiles for key-block 0
    if (threadIdx.x < 32) {
        tdm_load_2d_bf16(klds[0], Kp, HDIM, 32, HDIM, 32, HDIM);      // 32 keys x 64d
        tdm_load_2d_bf16(vlds[0], Vp, 32, HDIM, 32, HDIM, LL);        // 64d x 32 keys
    }

    // Q A-fragments for the two k-steps of Dh=64
    v16bf qf[2];
#pragma unroll
    for (int s = 0; s < 2; s++) {
        const __bf16* p = Qp + (size_t)(q0 + l16) * HDIM + s * 32 + lh * 16;
        *(v8bf*)&qf[s]       = *(const v8bf*)p;
        *((v8bf*)&qf[s] + 1) = *(const v8bf*)(p + 8);
    }

    v8f   cacc[4] = {};                          // ctx accum over 4 Dh tiles
    float m[8], lsum[8];
#pragma unroll
    for (int r = 0; r < 8; r++) { m[r] = -INFINITY; lsum[r] = 0.f; }

    const float scale = 0.125f;                  // 1/sqrt(64)
    __bf16* pb = pbuf[wave];

    for (int kb = 0; kb < LL / 32; kb++) {
        const int kbase = kb * 32;
        if (threadIdx.x < 32) {
            if (kb + 1 < LL / 32) {
                tdm_load_2d_bf16(klds[(kb + 1) & 1], Kp + (size_t)(kbase + 32) * HDIM,
                                 HDIM, 32, HDIM, 32, HDIM);
                tdm_load_2d_bf16(vlds[(kb + 1) & 1], Vp + (kbase + 32),
                                 32, HDIM, 32, HDIM, LL);
                __builtin_amdgcn_s_wait_tensorcnt(2);
            } else {
                __builtin_amdgcn_s_wait_tensorcnt(0);
            }
        }
        __syncthreads();

        const __bf16* kbuf = &kst[kb & 1][0];
        const __bf16* vbuf = &vst[kb & 1][0];

        // scores S[16 x 32] as two 16x16 tiles, f32 accumulate
        v8f s0 = {}, s1 = {};
#pragma unroll
        for (int s = 0; s < 2; s++) {
            const __bf16* pk = kbuf + (size_t)l16 * HDIM + s * 32 + lh * 16;
            v16bf b0 = *(const v16bf*)pk;
            v16bf b1 = *(const v16bf*)(pk + 16 * HDIM);
            s0 = __builtin_amdgcn_wmma_f32_16x16x32_bf16(false, qf[s], false, b0, (short)0, s0, false, false);
            s1 = __builtin_amdgcn_wmma_f32_16x16x32_bf16(false, qf[s], false, b1, (short)0, s1, false, false);
        }

        float p0a[8], p1a[8];
#pragma unroll
        for (int r = 0; r < 8; r++) {
            float v = fmaxf(s0[r], s1[r]);
#pragma unroll
            for (int off = 1; off < 16; off <<= 1) v = fmaxf(v, __shfl_xor(v, off, 32));
            const float mnew = fmaxf(m[r], v * scale);
            const float corr = __expf(m[r] - mnew);
            m[r] = mnew;
            const float p0 = __expf(s0[r] * scale - mnew);
            const float p1 = __expf(s1[r] * scale - mnew);
            p0a[r] = p0; p1a[r] = p1;
            float ps = p0 + p1;
#pragma unroll
            for (int off = 1; off < 16; off <<= 1) ps += __shfl_xor(ps, off, 32);
            lsum[r] = lsum[r] * corr + ps;
#pragma unroll
            for (int t = 0; t < 4; t++) cacc[t][r] *= corr;
        }

        // P (C-layout) -> LDS row-major [16][32] -> reload as A-fragment
#pragma unroll
        for (int r = 0; r < 8; r++) {
            const int row = r + 8 * lh;
            pb[row * 32 + l16]      = f2bf(p0a[r]);
            pb[row * 32 + 16 + l16] = f2bf(p1a[r]);
        }
        asm volatile("s_wait_dscnt 0" ::: "memory");
        v16bf pf;
        {
            const __bf16* pl = pb + (size_t)l16 * 32 + lh * 16;
            *(v8bf*)&pf       = *(const v8bf*)pl;
            *((v8bf*)&pf + 1) = *(const v8bf*)(pl + 8);
        }

        // ctx += P @ V  (B from staged Vt tile)
#pragma unroll
        for (int t = 0; t < 4; t++) {
            const __bf16* pv = vbuf + (size_t)(t * 16 + l16) * 32 + lh * 16;
            v16bf bv = *(const v16bf*)pv;
            cacc[t] = __builtin_amdgcn_wmma_f32_16x16x32_bf16(false, pf, false, bv, (short)0, cacc[t], false, false);
        }
        __syncthreads();
    }

    // normalize and store merged heads [B*L, 1024] bf16
    const int b = bh >> 4, h = bh & 15;
#pragma unroll
    for (int r = 0; r < 8; r++) {
        const float inv = 1.0f / lsum[r];
        const int row = q0 + r + 8 * lh;
#pragma unroll
        for (int t = 0; t < 4; t++) {
            const int d = h * HDIM + t * 16 + l16;
            ctx[((size_t)b * LL + row) * D_MODEL + d] = f2bf(cacc[t][r] * inv);
        }
    }
}

// ---------- launcher ----------
extern "C" void kernel_launch(void* const* d_in, const int* in_sizes, int n_in,
                              void* d_out, int out_size, void* d_ws, size_t ws_size,
                              hipStream_t stream) {
    (void)in_sizes; (void)n_in; (void)out_size; (void)ws_size;
    const float* q  = (const float*)d_in[0];
    const float* k  = (const float*)d_in[1];
    const float* v  = (const float*)d_in[2];
    const float* Wq = (const float*)d_in[3];
    const float* bq = (const float*)d_in[4];
    const float* Wk = (const float*)d_in[5];
    const float* bk = (const float*)d_in[6];
    const float* Wv = (const float*)d_in[7];
    const float* bv = (const float*)d_in[8];
    const float* Wo = (const float*)d_in[9];
    const float* bo = (const float*)d_in[10];

    char* ws = (char*)d_ws;
    const size_t MB = 1024 * 1024;
    __bf16* qx  = (__bf16*)(ws + 0 * MB);    // 8MB each input cast
    __bf16* kx  = (__bf16*)(ws + 8 * MB);
    __bf16* vx  = (__bf16*)(ws + 16 * MB);
    __bf16* wqb = (__bf16*)(ws + 24 * MB);   // 2MB each weight cast
    __bf16* wkb = (__bf16*)(ws + 26 * MB);
    __bf16* wvb = (__bf16*)(ws + 28 * MB);
    __bf16* wob = (__bf16*)(ws + 30 * MB);
    __bf16* Qh  = (__bf16*)(ws + 32 * MB);   // [B,H,L,64]
    __bf16* Kh  = (__bf16*)(ws + 40 * MB);   // [B,H,L,64]
    __bf16* Vt  = (__bf16*)(ws + 48 * MB);   // [B,H,64,L]
    __bf16* ctx = (__bf16*)(ws + 56 * MB);   // [B*L, 1024]

    const int nx = NTOK * D_MODEL;           // 4M
    const int nw = D_MODEL * D_MODEL;        // 1M
    cast_f32_bf16<<<nx / 256, 256, 0, stream>>>(q,  qx,  nx);
    cast_f32_bf16<<<nx / 256, 256, 0, stream>>>(k,  kx,  nx);
    cast_f32_bf16<<<nx / 256, 256, 0, stream>>>(v,  vx,  nx);
    cast_f32_bf16<<<nw / 256, 256, 0, stream>>>(Wq, wqb, nw);
    cast_f32_bf16<<<nw / 256, 256, 0, stream>>>(Wk, wkb, nw);
    cast_f32_bf16<<<nw / 256, 256, 0, stream>>>(Wv, wvb, nw);
    cast_f32_bf16<<<nw / 256, 256, 0, stream>>>(Wo, wob, nw);

    dim3 ggrid(D_MODEL / 128, NTOK / 128);   // (8, 32)
    gemm_nt_bf16<<<ggrid, 128, 0, stream>>>(qx,  wqb, bq, (void*)Qh, 1);
    gemm_nt_bf16<<<ggrid, 128, 0, stream>>>(kx,  wkb, bk, (void*)Kh, 1);
    gemm_nt_bf16<<<ggrid, 128, 0, stream>>>(vx,  wvb, bv, (void*)Vt, 2);

    attn_kernel<<<NTOK / 4, 128, 0, stream>>>(Qh, Kh, Vt, ctx);

    gemm_nt_bf16<<<ggrid, 128, 0, stream>>>(ctx, wob, bo, d_out, 0);
}
